// GroupQueryAttn_45664092291681
// MI455X (gfx1250) — matrix operand order repaired and checked by reading, not compiled, not run on hardware
//
#include <hip/hip_runtime.h>
#include <hip/hip_bf16.h>

// ---------------------------------------------------------------------------
// Types
// ---------------------------------------------------------------------------
typedef unsigned short u16;
typedef __attribute__((ext_vector_type(16))) __bf16 v16bf;
typedef __attribute__((ext_vector_type(8)))  float  v8f;

union Frag16 {
    v16bf v;
    uint4 q[2];
};

__device__ __forceinline__ u16 f2bf(float f) {
    union { float f; unsigned u; } x; x.f = f;
    unsigned u = x.u;
    u += 0x7fffu + ((u >> 16) & 1u);   // round-to-nearest-even
    return (u16)(u >> 16);
}

// ---------------------------------------------------------------------------
// fp32 -> bf16 cast kernel
// ---------------------------------------------------------------------------
__global__ __launch_bounds__(256)
void cast_f32_bf16(const float* __restrict__ in, u16* __restrict__ out, long n) {
    long i = (long)blockIdx.x * 256 + threadIdx.x;
    long stride = (long)gridDim.x * 256;
    for (; i < n; i += stride) out[i] = f2bf(in[i]);
}

// ---------------------------------------------------------------------------
// bf16 GEMM: C[M,N] = A[M,K] * B[K,N] + bias[N]
//   A, B row-major bf16.  F32OUT ? fp32 out : bf16 out.
//   Block tile 128x128, BK=64, 256 threads (8 waves), wave tile 64x32.
// ---------------------------------------------------------------------------
#define GBM 128
#define GBN 128
#define GBK 64
#define GLS 72   // GBK + 8 pad (ushort elements), keeps 16B alignment

template<bool F32OUT>
__global__ __launch_bounds__(256)
void gemm_bf16(const u16* __restrict__ A, const u16* __restrict__ B,
               const float* __restrict__ bias, void* __restrict__ Cout,
               int M, int N, int K)
{
    __shared__ __align__(16) u16 sA[GBM * GLS];   // [m][k]
    __shared__ __align__(16) u16 sB[GBN * GLS];   // [n][k]  (transposed)

    const int tid  = threadIdx.x;
    const int wave = tid >> 5;
    const int lane = tid & 31;
    const int l15  = lane & 15;
    const int lh   = lane >> 4;

    const int m0 = blockIdx.y * GBM;
    const int n0 = blockIdx.x * GBN;
    const int wm = wave >> 2;   // 0..1 -> 64-row strip
    const int wn = wave & 3;    // 0..3 -> 32-col strip

    v8f acc[4][2];
#pragma unroll
    for (int mt = 0; mt < 4; ++mt)
#pragma unroll
        for (int nt = 0; nt < 2; ++nt) acc[mt][nt] = (v8f)0.0f;

    for (int kb = 0; kb < K; kb += GBK) {
        // ---- A tile: 128x64, 1024 x 8-elem chunks ----
#pragma unroll
        for (int c = tid; c < (GBM * GBK / 8); c += 256) {
            const int i  = c >> 3;
            const int kc = (c & 7) * 8;
            *(uint4*)&sA[i * GLS + kc] =
                *(const uint4*)&A[(size_t)(m0 + i) * K + kb + kc];
        }
        // ---- B tile transposed into [n][k] ----
#pragma unroll
        for (int c = tid; c < (GBK * GBN / 8); c += 256) {
            const int kk = c >> 4;
            const int nc = (c & 15) * 8;
            union { uint4 q; u16 s[8]; } u;
            u.q = *(const uint4*)&B[(size_t)(kb + kk) * N + n0 + nc];
#pragma unroll
            for (int j = 0; j < 8; ++j)
                sB[(nc + j) * GLS + kk] = u.s[j];
        }
        if (kb + GBK < K) {
            __builtin_prefetch(&A[(size_t)(m0 + (tid >> 1)) * K + kb + GBK], 0, 1);
            __builtin_prefetch(&B[(size_t)(kb + GBK + (tid >> 2)) * N + n0], 0, 1);
        }
        __syncthreads();

#pragma unroll
        for (int k0 = 0; k0 < GBK; k0 += 32) {
            Frag16 fa[4], fb[2];
#pragma unroll
            for (int mt = 0; mt < 4; ++mt) {
                const u16* p = &sA[(wm * 64 + mt * 16 + l15) * GLS + k0 + lh * 8];
                fa[mt].q[0] = *(const uint4*)p;
                fa[mt].q[1] = *(const uint4*)(p + 16);
            }
#pragma unroll
            for (int nt = 0; nt < 2; ++nt) {
                const u16* p = &sB[(wn * 32 + nt * 16 + l15) * GLS + k0 + lh * 8];
                fb[nt].q[0] = *(const uint4*)p;
                fb[nt].q[1] = *(const uint4*)(p + 16);
            }
#pragma unroll
            for (int mt = 0; mt < 4; ++mt)
#pragma unroll
                for (int nt = 0; nt < 2; ++nt)
                    acc[mt][nt] = __builtin_amdgcn_wmma_f32_16x16x32_bf16(
                        false, fa[mt].v, false, fb[nt].v,
                        (short)0, acc[mt][nt], false, false);
        }
        __syncthreads();
    }

    // ---- epilogue: C layout: lane holds col n=l15, rows r + 8*lh ----
#pragma unroll
    for (int mt = 0; mt < 4; ++mt)
#pragma unroll
        for (int nt = 0; nt < 2; ++nt) {
            const int gcol = n0 + wn * 32 + nt * 16 + l15;
            const float bv = bias[gcol];
#pragma unroll
            for (int r = 0; r < 8; ++r) {
                const int grow = m0 + wm * 64 + mt * 16 + r + lh * 8;
                const float v = acc[mt][nt][r] + bv;
                if (F32OUT)
                    ((float*)Cout)[(size_t)grow * N + gcol] = v;
                else
                    ((u16*)Cout)[(size_t)grow * N + gcol] = f2bf(v);
            }
        }
}

// ---------------------------------------------------------------------------
// Flash attention (GQA): H=32 heads, G=8 groups (4 heads/group), D=128.
//   One block per (q-tile of 64 rows, head, batch); 128 threads = 4 waves,
//   each wave owns 16 query rows. Online softmax, bf16 WMMA, fp32 accum.
// ---------------------------------------------------------------------------
#define AQ 64
#define AK 64
#define DH 128
#define QS 136   // DH + 8
#define KS 136
#define VS 72    // AK + 8
#define PS 72

__global__ __launch_bounds__(128)
void attn_gqa(const u16* __restrict__ Qg, const u16* __restrict__ Kg,
              const u16* __restrict__ Vg, u16* __restrict__ Mid, int S)
{
    __shared__ __align__(16) u16 sQ [AQ * QS];       // [qrow][d]
    __shared__ __align__(16) u16 sK [AK * KS];       // [key][d]   == B^T for QK^T
    __shared__ __align__(16) u16 sVt[DH * VS];       // [d][key]   == B^T for PV
    __shared__ __align__(16) u16 sP [4 * 16 * PS];   // per-wave P stage [16][keys]

    const int tid  = threadIdx.x;
    const int wq   = tid >> 5;
    const int lane = tid & 31;
    const int l15  = lane & 15;
    const int lh   = lane >> 4;

    const int qt = blockIdx.x, h = blockIdx.y, b = blockIdx.z;
    const int g  = h >> 2;                         // H/G = 4 heads per group
    const size_t tok0 = (size_t)b * S + (size_t)qt * AQ;

    const u16* Qp = Qg + tok0 * 4096 + (size_t)h * DH;
    const u16* Kp = Kg + (size_t)b * S * 1024 + (size_t)g * DH;
    const u16* Vp = Vg + (size_t)b * S * 1024 + (size_t)g * DH;

    // load Q tile 64x128
#pragma unroll
    for (int c = tid; c < AQ * (DH / 8); c += 128) {
        const int i = c >> 4, dc = (c & 15) * 8;
        *(uint4*)&sQ[i * QS + dc] = *(const uint4*)&Qp[(size_t)i * 4096 + dc];
    }

    v8f acc[8];
#pragma unroll
    for (int dt = 0; dt < 8; ++dt) acc[dt] = (v8f)0.0f;
    float mi[8], li[8];
#pragma unroll
    for (int r = 0; r < 8; ++r) { mi[r] = -1e30f; li[r] = 0.0f; }

    const float c2 = 0.08838834764831845f * 1.4426950408889634f; // 1/sqrt(D)*log2(e)
    u16* sPw = &sP[wq * 16 * PS];

    for (int kt = 0; kt < S / AK; ++kt) {
        __syncthreads();   // previous tile's reads done before overwrite
        const u16* Kt = Kp + (size_t)kt * AK * 1024;
        const u16* Vt = Vp + (size_t)kt * AK * 1024;
        // K tile row-major
#pragma unroll
        for (int c = tid; c < AK * (DH / 8); c += 128) {
            const int i = c >> 4, dc = (c & 15) * 8;
            *(uint4*)&sK[i * KS + dc] = *(const uint4*)&Kt[(size_t)i * 1024 + dc];
        }
        // V tile transposed -> [d][key]
#pragma unroll
        for (int c = tid; c < AK * (DH / 8); c += 128) {
            const int i = c >> 4, dc = (c & 15) * 8;
            union { uint4 q; u16 s[8]; } u;
            u.q = *(const uint4*)&Vt[(size_t)i * 1024 + dc];
#pragma unroll
            for (int j = 0; j < 8; ++j) sVt[(dc + j) * VS + i] = u.s[j];
        }
        if (kt + 1 < S / AK)
            __builtin_prefetch(&Kt[(size_t)(AK + (tid >> 1)) * 1024], 0, 1);
        __syncthreads();

        // ---- scores: S = Q * K^T, 16x64 per wave ----
        v8f sv[4];
#pragma unroll
        for (int nt = 0; nt < 4; ++nt) {
            sv[nt] = (v8f)0.0f;
#pragma unroll
            for (int k0 = 0; k0 < DH; k0 += 32) {
                Frag16 fa, fb;
                const u16* pa = &sQ[(wq * 16 + l15) * QS + k0 + lh * 8];
                fa.q[0] = *(const uint4*)pa;
                fa.q[1] = *(const uint4*)(pa + 16);
                const u16* pb = &sK[(nt * 16 + l15) * KS + k0 + lh * 8];
                fb.q[0] = *(const uint4*)pb;
                fb.q[1] = *(const uint4*)(pb + 16);
                sv[nt] = __builtin_amdgcn_wmma_f32_16x16x32_bf16(
                    false, fa.v, false, fb.v, (short)0, sv[nt], false, false);
            }
        }

        // ---- online softmax (row = r + 8*lh, spread over 16 lanes) ----
#pragma unroll
        for (int r = 0; r < 8; ++r) {
            float rm = fmaxf(fmaxf(sv[0][r], sv[1][r]), fmaxf(sv[2][r], sv[3][r]));
#pragma unroll
            for (int off = 8; off; off >>= 1) rm = fmaxf(rm, __shfl_xor(rm, off, 32));
            rm *= c2;
            const float mnew = fmaxf(mi[r], rm);
            const float corr = __builtin_amdgcn_exp2f(mi[r] - mnew);
            float rs = 0.0f;
#pragma unroll
            for (int nt = 0; nt < 4; ++nt) {
                const float p = __builtin_amdgcn_exp2f(sv[nt][r] * c2 - mnew);
                rs += p;
                sPw[(r + 8 * lh) * PS + nt * 16 + l15] = f2bf(p);
            }
#pragma unroll
            for (int off = 8; off; off >>= 1) rs += __shfl_xor(rs, off, 32);
            li[r] = li[r] * corr + rs;
            mi[r] = mnew;
#pragma unroll
            for (int dt = 0; dt < 8; ++dt) acc[dt][r] *= corr;
        }

        // ---- O += P * V ----
#pragma unroll
        for (int k0 = 0; k0 < AK; k0 += 32) {
            Frag16 fa;
            const u16* pa = &sPw[l15 * PS + k0 + lh * 8];
            fa.q[0] = *(const uint4*)pa;
            fa.q[1] = *(const uint4*)(pa + 16);
#pragma unroll
            for (int dt = 0; dt < 8; ++dt) {
                Frag16 fb;
                const u16* pb = &sVt[(dt * 16 + l15) * VS + k0 + lh * 8];
                fb.q[0] = *(const uint4*)pb;
                fb.q[1] = *(const uint4*)(pb + 16);
                acc[dt] = __builtin_amdgcn_wmma_f32_16x16x32_bf16(
                    false, fa.v, false, fb.v, (short)0, acc[dt], false, false);
            }
        }
    }

    // ---- normalize and write mid[token][h*128 + d] as bf16 ----
    float inv[8];
#pragma unroll
    for (int r = 0; r < 8; ++r) inv[r] = (li[r] > 0.0f) ? (1.0f / li[r]) : 0.0f;
    u16* Op = Mid + tok0 * 4096 + (size_t)h * DH;
#pragma unroll
    for (int dt = 0; dt < 8; ++dt)
#pragma unroll
        for (int r = 0; r < 8; ++r) {
            const int row = wq * 16 + r + 8 * lh;
            const int col = dt * 16 + l15;
            Op[(size_t)row * 4096 + col] = f2bf(acc[dt][r] * inv[r]);
        }
}

// ---------------------------------------------------------------------------
// Host-side orchestration
// ---------------------------------------------------------------------------
extern "C" void kernel_launch(void* const* d_in, const int* in_sizes, int n_in,
                              void* d_out, int out_size, void* d_ws, size_t ws_size,
                              hipStream_t stream) {
    (void)in_sizes; (void)n_in; (void)out_size; (void)ws_size;

    constexpr int B = 2, S = 2048, HID = 4096, GD = 1024, H = 32;
    constexpr long NX = (long)B * S * HID;       // 16,777,216
    constexpr long NW = (long)HID * HID;         // 16,777,216
    constexpr long NWK = (long)HID * GD;         // 4,194,304

    const float* X  = (const float*)d_in[0];
    const float* Wq = (const float*)d_in[1];
    const float* bq = (const float*)d_in[2];
    const float* Wk = (const float*)d_in[3];
    const float* bk = (const float*)d_in[4];
    const float* Wv = (const float*)d_in[5];
    const float* bv = (const float*)d_in[6];
    const float* Wo = (const float*)d_in[7];
    const float* bo = (const float*)d_in[8];
    float* out = (float*)d_out;

    // workspace carve-up (bf16 elements)
    u16* w   = (u16*)d_ws;
    u16* Xb  = w;              w += NX;
    u16* Wqb = w;              w += NW;
    u16* Wkb = w;              w += NWK;
    u16* Wvb = w;              w += NWK;
    u16* Wob = w;              w += NW;
    u16* Qb  = w;              w += NX;
    u16* Kb  = w;              w += (long)B * S * GD;
    u16* Vb  = w;              w += (long)B * S * GD;
    u16* Mb  = w;              w += NX;

    // 1) fp32 -> bf16 casts
    cast_f32_bf16<<<8192, 256, 0, stream>>>(X,  Xb,  NX);
    cast_f32_bf16<<<8192, 256, 0, stream>>>(Wq, Wqb, NW);
    cast_f32_bf16<<<4096, 256, 0, stream>>>(Wk, Wkb, NWK);
    cast_f32_bf16<<<4096, 256, 0, stream>>>(Wv, Wvb, NWK);
    cast_f32_bf16<<<8192, 256, 0, stream>>>(Wo, Wob, NW);

    const int M = B * S;   // 4096 tokens

    // 2) Q/K/V projections (bf16 out)
    gemm_bf16<false><<<dim3(HID / GBN, M / GBM), 256, 0, stream>>>(
        Xb, Wqb, bq, Qb, M, HID, HID);
    gemm_bf16<false><<<dim3(GD / GBN, M / GBM), 256, 0, stream>>>(
        Xb, Wkb, bk, Kb, M, GD, HID);
    gemm_bf16<false><<<dim3(GD / GBN, M / GBM), 256, 0, stream>>>(
        Xb, Wvb, bv, Vb, M, GD, HID);

    // 3) flash attention
    attn_gqa<<<dim3(S / AQ, H, B), 128, 0, stream>>>(Qb, Kb, Vb, Mb, S);

    // 4) output projection (fp32 out + bias)
    gemm_bf16<true><<<dim3(HID / GBN, M / GBM), 256, 0, stream>>>(
        Mb, Wob, bo, out, M, HID, HID);
}
